// HybridQLSTM_65481071395752
// MI455X (gfx1250) — matrix operand order repaired and checked by reading, not compiled, not run on hardware
//
#include <hip/hip_runtime.h>
#include <math.h>

// ---------------------------------------------------------------------------
// LSTM  T=1024  B=64  D_IN=256  D_H=256   (gates N = 4H = 1024, K = 512)
//
// Phase 0: pack  W -> Wx_f16 / Wh_f16 (n-major == WMMA-B-friendly), bias, X->f16
// Phase 1: parallel WMMA GEMM  gx[65536,1024] = X_f16 @ Wx^T + bias  (f32)
// Phase 2: single 1024-thread workgroup (32 wave32s, one WGP) runs the whole
//          recurrence. Per step: 64x1024x256 WMMA GEMM with A(=h) in LDS,
//          B(=Wh) streamed from L2, C initialized from gx[t]; activations
//          staged in a 256 KB f32 LDS buffer (CDNA5 320 KB WGP LDS);
//          then the cross-gate elementwise c/h update. gx[t+1] prefetched.
// ---------------------------------------------------------------------------

typedef __attribute__((ext_vector_type(16))) _Float16 v16h;
typedef __attribute__((ext_vector_type(8)))  _Float16 h8;
typedef __attribute__((ext_vector_type(8)))  float    v8f;

#define TSTEPS 1024
#define BATCH  64
#define DH     256
#define N4H    1024
#define MROWS  (TSTEPS * BATCH)   // 65536

__device__ __forceinline__ float sig_fast(float x)  { return 1.0f / (1.0f + __expf(-x)); }
__device__ __forceinline__ float tanh_fast(float x) { return 2.0f / (1.0f + __expf(-2.0f * x)) - 1.0f; }

// ---------------- Phase 0: pack weights / bias / X to f16 ------------------
__global__ void lstm_pack_kernel(const float* __restrict__ Wf, const float* __restrict__ Wi,
                                 const float* __restrict__ Wg, const float* __restrict__ Wo,
                                 const float* __restrict__ bf, const float* __restrict__ bi,
                                 const float* __restrict__ bg, const float* __restrict__ bo,
                                 const float* __restrict__ X,
                                 _Float16* __restrict__ Xh, _Float16* __restrict__ Wx,
                                 _Float16* __restrict__ Wh, float* __restrict__ bias) {
    size_t id = (size_t)blockIdx.x * blockDim.x + threadIdx.x;
    size_t stride = (size_t)gridDim.x * blockDim.x;
    const float* Ws[4] = {Wf, Wi, Wg, Wo};
    const float* bs[4] = {bf, bi, bg, bo};
    for (size_t i = id; i < (size_t)N4H * 512; i += stride) {
        int n = (int)(i >> 9), k = (int)(i & 511);
        float v = Ws[n >> 8][(size_t)(n & 255) * 512 + k];
        if (k < 256) Wx[(size_t)n * 256 + k]         = (_Float16)v;
        else         Wh[(size_t)n * 256 + (k - 256)] = (_Float16)v;
    }
    for (size_t i = id; i < N4H; i += stride) bias[i] = bs[i >> 8][i & 255];
    for (size_t i = id; i < (size_t)MROWS * 256; i += stride) Xh[i] = (_Float16)X[i];
}

// A operand (16x32 f16) from row-major storage (ld halves/row).
// lanes 0-15 = rows; slots 0-7 -> K = k0+8*hh .. +7; slots 8-15 -> K = k0+16+8*hh .. +7
__device__ __forceinline__ v16h load_a16x32(const _Float16* __restrict__ base,
                                            int ld, int ln, int hh, int k0) {
    const _Float16* p = base + (size_t)ln * ld + k0 + 8 * hh;
    h8 lo = *(const h8*)(p);
    h8 hi = *(const h8*)(p + 16);
    return __builtin_shufflevector(lo, hi, 0,1,2,3,4,5,6,7,8,9,10,11,12,13,14,15);
}

// B operand (32x16 f16) from n-major weights W[n][k]:
// per lane: col n = ln, slots 0-7 -> K = k0+8*hh .. +7; slots 8-15 -> K = k0+16+8*hh .. +7
__device__ __forceinline__ v16h load_b32x16(const _Float16* __restrict__ W,
                                            int ld, int ln, int hh, int n0, int k0) {
    const _Float16* p = W + (size_t)(n0 + ln) * ld + k0 + 8 * hh;
    h8 lo = *(const h8*)(p);
    h8 hi = *(const h8*)(p + 16);
    return __builtin_shufflevector(lo, hi, 0,1,2,3,4,5,6,7,8,9,10,11,12,13,14,15);
}

// ---------------- Phase 1: gx = X_f16 @ Wx^T + bias  (WMMA) ----------------
// 8 waves/block; wave = (m-tile, n-group of 8 tiles). A preloaded (8x v16h),
// j-outer so only one accumulator is live -> B loads pipeline across WMMAs.
__global__ __launch_bounds__(256) void lstm_xproj_kernel(
        const _Float16* __restrict__ Xh, const _Float16* __restrict__ Wx,
        const float* __restrict__ bias, float* __restrict__ gx) {
    int wid  = blockIdx.x * 8 + (threadIdx.x >> 5);
    int lane = threadIdx.x & 31;
    int ln = lane & 15, hh = lane >> 4;
    int mt  = wid >> 3;          // 0..2047
    int ntg = wid & 7;           // 0..7
    int m0 = mt * 16, n0g = ntg * 128;

    v16h a[8];
#pragma unroll
    for (int kt = 0; kt < 8; ++kt)
        a[kt] = load_a16x32(Xh + (size_t)m0 * 256, 256, ln, hh, kt * 32);

#pragma unroll
    for (int j = 0; j < 8; ++j) {
        int n = n0g + j * 16 + ln;
        v8f acc = {};
#pragma unroll
        for (int kt = 0; kt < 8; ++kt) {
            v16h b = load_b32x16(Wx, 256, ln, hh, n0g + j * 16, kt * 32);
            acc = __builtin_amdgcn_wmma_f32_16x16x32_f16(
                false, a[kt], false, b, (short)0, acc, false, false);
        }
        float bv = bias[n];
#pragma unroll
        for (int r = 0; r < 8; ++r)
            gx[(size_t)(m0 + r + 8 * hh) * N4H + n] = acc[r] + bv;
    }
}

// ---------------- Phase 2: fused sequential LSTM (one workgroup) -----------
__global__ __launch_bounds__(1024) void lstm_seq_kernel(
        const _Float16* __restrict__ Wh, const float* __restrict__ gx,
        const float* __restrict__ h0, const float* __restrict__ c0,
        float* __restrict__ out) {
    __shared__ _Float16 hS[BATCH * DH];      // 32 KB, h_t (f16, row-major)
    __shared__ float    actS[BATCH * N4H];   // 256 KB, activated gates (f32)

    int tid = threadIdx.x;
    int w = tid >> 5, lane = tid & 31;
    int ln = lane & 15, hh = lane >> 4;
    int mt = w >> 3, ntg = w & 7;            // 4 m-tiles x 8 n-groups = 32 waves
    int m0 = mt * 16, n0g = ntg * 128;
    int gate = ntg >> 1;                     // 0=f 1=i 2=g 3=o

    float c[16];
#pragma unroll
    for (int r = 0; r < 16; ++r) {
        int e = r * 1024 + tid;              // covers 0..16383
        c[r] = c0[e];
        hS[e] = (_Float16)h0[e];
    }
    __syncthreads();

    float* hout = out + (size_t)TSTEPS * BATCH * DH;
    float* cout = hout + BATCH * DH;

    for (int t = 0; t < TSTEPS; ++t) {
        // A operands for this wave's m-tile (reused across all 8 n-tiles)
        v16h a[8];
#pragma unroll
        for (int kt = 0; kt < 8; ++kt)
            a[kt] = load_a16x32(hS + (size_t)m0 * DH, DH, ln, hh, kt * 32);

        // prefetch next step's gx slab (256 KB) into L2 while we compute
        if (t + 1 < TSTEPS) {
            const float* nx = gx + (size_t)(t + 1) * BATCH * N4H + (size_t)tid * 64;
            __builtin_prefetch(nx, 0, 0);
            __builtin_prefetch(nx + 32, 0, 0);
        }

        const float* gxt = gx + (size_t)t * BATCH * N4H;
#pragma unroll
        for (int j = 0; j < 8; ++j) {
            int n = n0g + j * 16 + ln;
            // C operand initialized from precomputed x-projection (+bias)
            v8f acc;
#pragma unroll
            for (int r = 0; r < 8; ++r)
                acc[r] = gxt[(size_t)(m0 + r + 8 * hh) * N4H + n];
#pragma unroll
            for (int kt = 0; kt < 8; ++kt) {
                v16h b = load_b32x16(Wh, 256, ln, hh, n0g + j * 16, kt * 32);
                acc = __builtin_amdgcn_wmma_f32_16x16x32_f16(
                    false, a[kt], false, b, (short)0, acc, false, false);
            }
#pragma unroll
            for (int r = 0; r < 8; ++r) {
                float v = acc[r];
                v = (gate == 2) ? tanh_fast(v) : sig_fast(v);
                actS[(m0 + r + 8 * hh) * N4H + n] = v;
            }
        }
        __syncthreads();
        // cross-gate elementwise state update (16 elems / thread)
        float* orow = out + (size_t)t * (BATCH * DH);
#pragma unroll
        for (int r = 0; r < 16; ++r) {
            int e = r * 1024 + tid;
            int b = e >> 8, j = e & 255;
            float fg = actS[b * N4H + j];
            float ig = actS[b * N4H + 256 + j];
            float gg = actS[b * N4H + 512 + j];
            float og = actS[b * N4H + 768 + j];
            float cn = fg * c[r] + ig * gg;
            c[r] = cn;
            float hn = og * tanh_fast(cn);
            orow[e] = hn;
            hS[b * DH + j] = (_Float16)hn;   // feeds next step's A operand
            if (t == TSTEPS - 1) { hout[e] = hn; cout[e] = cn; }
        }
        __syncthreads();
    }
}

// ---------------------------------------------------------------------------
extern "C" void kernel_launch(void* const* d_in, const int* in_sizes, int n_in,
                              void* d_out, int out_size, void* d_ws, size_t ws_size,
                              hipStream_t stream) {
    const float* X  = (const float*)d_in[0];
    const float* h0 = (const float*)d_in[1];
    const float* c0 = (const float*)d_in[2];
    const float* Wf = (const float*)d_in[3];  const float* bf = (const float*)d_in[4];
    const float* Wi = (const float*)d_in[5];  const float* bi = (const float*)d_in[6];
    const float* Wg = (const float*)d_in[7];  const float* bg = (const float*)d_in[8];
    const float* Wo = (const float*)d_in[9];  const float* bo = (const float*)d_in[10];

    char* ws = (char*)d_ws;
    _Float16* Xh   = (_Float16*)(ws);                               // 32 MB
    _Float16* Wx   = (_Float16*)(ws + (size_t)33554432);            // 512 KB
    _Float16* Wh   = (_Float16*)(ws + (size_t)33554432 + 524288);   // 512 KB
    float*    bias = (float*)   (ws + (size_t)33554432 + 1048576);  // 4 KB
    float*    gx   = (float*)   (ws + (size_t)33554432 + 1048576 + 4096); // 256 MB

    lstm_pack_kernel<<<1024, 256, 0, stream>>>(Wf, Wi, Wg, Wo, bf, bi, bg, bo,
                                               X, Xh, Wx, Wh, bias);
    lstm_xproj_kernel<<<MROWS / 16 / 2, 256, 0, stream>>>(Xh, Wx, bias, gx);
    lstm_seq_kernel<<<1, 1024, 0, stream>>>(Wh, gx, h0, c0, (float*)d_out);
}
